// Attention_25254407701324
// MI455X (gfx1250) — matrix operand order repaired and checked by reading, not compiled, not run on hardware
//
#include <hip/hip_runtime.h>
#include <hip/hip_bf16.h>

#define DEV_INLINE __device__ __forceinline__

typedef __bf16 bf16;
typedef __attribute__((ext_vector_type(16))) __bf16 v16bf;
typedef __attribute__((ext_vector_type(8)))  __bf16 v8bf;
typedef __attribute__((ext_vector_type(8)))  float  v8f;

// Problem constants (fixed by the reference)
constexpr int Bc = 2, Nc = 2048, Ec = 1024, Hc = 16, Dc = 64;
constexpr int Mrows = Bc * Nc;          // 4096

DEV_INLINE v8bf ld8(const bf16* p) { return *(const v8bf*)p; }

DEV_INLINE v16bf cat8(v8bf lo, v8bf hi) {
  v16bf r;
#pragma unroll
  for (int i = 0; i < 8; ++i) { r[i] = lo[i]; r[i + 8] = hi[i]; }
  return r;
}

DEV_INLINE v8f wmma_bf16(v16bf a, v16bf b, v8f c) {
  // D = A(16x32 bf16) * B(32x16 bf16) + C(16x16 f32)
  return __builtin_amdgcn_wmma_f32_16x16x32_bf16(
      /*neg_a=*/false, a, /*neg_b=*/false, b,
      /*c_mod=*/(short)0, c, /*reuse_a=*/false, /*reuse_b=*/false);
}

// CDNA5 async global->LDS copy, 16B per lane (tracked by ASYNCcnt).
DEV_INLINE void async_ld_b128(unsigned lds_off, const bf16* g) {
  asm volatile("global_load_async_to_lds_b128 %0, %1, off"
               :
               : "v"(lds_off), "v"(g)
               : "memory");
}

template <int N>
DEV_INLINE void wait_asynccnt() {
#if __has_builtin(__builtin_amdgcn_s_wait_asynccnt)
  __builtin_amdgcn_s_wait_asynccnt(N);
#else
  asm volatile("s_wait_asynccnt %0" : : "n"(N) : "memory");
#endif
}

// ---------------------------------------------------------------------------
// Kernel 1: f32 -> bf16 conversion (grid covers n elements; n % 256 == 0 here)
// ---------------------------------------------------------------------------
__global__ void cvt_bf16_kernel(const float* __restrict__ s,
                                bf16* __restrict__ d, int n) {
  int i = blockIdx.x * blockDim.x + threadIdx.x;
  if (i < n) d[i] = (bf16)s[i];
}

// ---------------------------------------------------------------------------
// Kernel 2: C[m,c] = sum_k A[m,k] * W[c,k] + bias[c]    (A @ W^T + b)
// A: [M,K] bf16 row-major, W: [NC,K] bf16 row-major.
// Block tile 128x128, K-step 32, 256 threads = 8 waves, each wave a 32x64
// sub-tile = 2x4 WMMA accumulators. Tiles staged into double-buffered LDS
// with global_load_async_to_lds_b128 so the next stage's fill overlaps the
// current stage's WMMAs (ASYNCcnt pipelining). LDS pitch 40 elems (80B rows).
// MODE 0: store bf16 into [B,H,N,D]   (Q and K, split-heads layout)
// MODE 1: store bf16 into [B,H,D,N]   (V transposed for the PV GEMM)
// MODE 2: store f32  into [M,NC]      (final output projection)
// ---------------------------------------------------------------------------
template <int MODE>
__global__ __launch_bounds__(256) void wmma_gemm_bias(
    const bf16* __restrict__ A, const bf16* __restrict__ W,
    const float* __restrict__ bias, void* __restrict__ dst,
    int M, int NCd, int K) {
  constexpr int LD = 40;                 // LDS pitch in elements (80B rows)
  __shared__ bf16 sA[2][128 * LD];
  __shared__ bf16 sB[2][128 * LD];

  const int tid = threadIdx.x;
  const int lane = tid & 31;
  const int w = tid >> 5;
  const int lr = lane & 15;              // row/col within a 16-tile
  const int hg = lane >> 4;              // half-group (0/1)
  const int wm = (w & 3) * 32;           // 4 waves along M
  const int wn = (w >> 2) * 64;          // 2 waves along N
  const int m0 = blockIdx.y * 128;
  const int n0 = blockIdx.x * 128;

  const bf16* gA = A + (size_t)m0 * K;
  const bf16* gW = W + (size_t)n0 * K;

  // Issue one stage: 128x32 A-tile + 128x32 B-tile as 16B async transfers.
  // 512 segments per tile, 256 threads -> 4 async ops per thread (ASYNCcnt+4
  // per wave per stage).
  auto stage = [&](int bufi, int k0) {
#pragma unroll
    for (int i = 0; i < 2; ++i) {
      int seg = tid + i * 256;           // 0..511
      int row = seg >> 2, sg = seg & 3;  // 4 x 16B per 32-elem row
      async_ld_b128((unsigned)(uintptr_t)&sA[bufi][row * LD + sg * 8],
                    &gA[(size_t)row * K + k0 + sg * 8]);
      async_ld_b128((unsigned)(uintptr_t)&sB[bufi][row * LD + sg * 8],
                    &gW[(size_t)row * K + k0 + sg * 8]);
    }
  };

  v8f acc[2][4];
#pragma unroll
  for (int mt = 0; mt < 2; ++mt)
#pragma unroll
    for (int nt = 0; nt < 4; ++nt) acc[mt][nt] = {};

  stage(0, 0);
  int buf = 0;
  for (int k0 = 0; k0 < K; k0 += 32) {
    __syncthreads();  // every wave is done reading buf^1 (previous compute)
    const bool has_next = (k0 + 32) < K;
    if (has_next) {
      stage(buf ^ 1, k0 + 32);
      wait_asynccnt<4>();   // own stage-k0 ops retired (in-order), next in flight
    } else {
      wait_asynccnt<0>();
    }
    __syncthreads();  // buf fully written by all waves

    // A fragments: lane holds row (lr), K in {8hg..8hg+7} u {16+8hg..}
    v16bf af[2];
#pragma unroll
    for (int mt = 0; mt < 2; ++mt) {
      int base = (wm + mt * 16 + lr) * LD + hg * 8;
      af[mt] = cat8(ld8(&sA[buf][base]), ld8(&sA[buf][base + 16]));
    }
    // B fragments: lane holds col (lr), K in {8hg..8hg+7} u {16+8hg..}
    v16bf bf_[4];
#pragma unroll
    for (int nt = 0; nt < 4; ++nt) {
      int base = (wn + nt * 16 + lr) * LD + hg * 8;
      bf_[nt] = cat8(ld8(&sB[buf][base]), ld8(&sB[buf][base + 16]));
    }
#pragma unroll
    for (int mt = 0; mt < 2; ++mt)
#pragma unroll
      for (int nt = 0; nt < 4; ++nt)
        acc[mt][nt] = wmma_bf16(af[mt], bf_[nt], acc[mt][nt]);

    buf ^= 1;
  }

  // Store: C layout -> lane = col (lr), reg r = row (8hg + r)
#pragma unroll
  for (int mt = 0; mt < 2; ++mt)
#pragma unroll
    for (int nt = 0; nt < 4; ++nt)
#pragma unroll
      for (int r = 0; r < 8; ++r) {
        int m = m0 + wm + mt * 16 + hg * 8 + r;
        int c = n0 + wn + nt * 16 + lr;
        float val = acc[mt][nt][r] + bias[c];
        if constexpr (MODE == 0) {
          int b = m >> 11, n = m & (Nc - 1);
          int h = c >> 6, d = c & (Dc - 1);
          ((bf16*)dst)[((size_t)(b * Hc + h) * Nc + n) * Dc + d] = (bf16)val;
        } else if constexpr (MODE == 1) {
          int b = m >> 11, n = m & (Nc - 1);
          int h = c >> 6, d = c & (Dc - 1);
          ((bf16*)dst)[((size_t)(b * Hc + h) * Dc + d) * Nc + n] = (bf16)val;
        } else {
          ((float*)dst)[(size_t)m * NCd + c] = val;
        }
      }
}

// ---------------------------------------------------------------------------
// Kernel 3: flash attention (softmax-then-1/sqrt(E), per the reference).
// 128 threads = 4 waves; each wave owns 16 queries of one (b,h).
// Computes S^T = K_tile(16x64) @ q^T per 16-key tile so exp(S) is already in
// the WMMA A-operand layout for the P@V GEMM (no cross-lane transpose).
// q/k/vT stream straight from global (all of q,k,v ~24MB lives in the 192MB
// L2), interleaving global_load_b128 clauses with WMMAs.
// ---------------------------------------------------------------------------
__global__ __launch_bounds__(128) void flash_attn_kernel(
    const bf16* __restrict__ q,   // [B,H,N,D]
    const bf16* __restrict__ k,   // [B,H,N,D]
    const bf16* __restrict__ vT,  // [B,H,D,N]
    bf16* __restrict__ attn) {    // [B*N, E] merged heads
  const int lane = threadIdx.x & 31;
  const int w = threadIdx.x >> 5;
  const int lr = lane & 15;
  const int hg = lane >> 4;

  const int bh = blockIdx.x >> 5;          // Nc/64 = 32 q-blocks per (b,h)
  const int qb = blockIdx.x & 31;
  const int qt = qb * 64 + w * 16;         // this wave's first query row
  const bf16* qp = q + (size_t)bh * Nc * Dc;
  const bf16* kp = k + (size_t)bh * Nc * Dc;
  const bf16* vp = vT + (size_t)bh * Dc * Nc;

  // q as WMMA B-operand (d is the K-dim): two K-steps of 32 over D=64.
  v16bf qf[2];
#pragma unroll
  for (int s = 0; s < 2; ++s) {
    int off = (qt + lr) * Dc + s * 32 + hg * 8;
    qf[s] = cat8(ld8(qp + off), ld8(qp + off + 16));
  }

  float mrun = -3.0e38f, lrun = 0.0f;
  v8f o[4];
#pragma unroll
  for (int dt = 0; dt < 4; ++dt) o[dt] = {};

  for (int kb0 = 0; kb0 < Nc; kb0 += 32) {
    // Two S^T tiles: (keys kb0..+15) x (16 queries), (keys +16..+31) x q.
    v8f st[2];
#pragma unroll
    for (int t = 0; t < 2; ++t) {
      v8f c = {};
#pragma unroll
      for (int s = 0; s < 2; ++s) {
        int off = (kb0 + t * 16 + lr) * Dc + s * 32 + hg * 8;
        v16bf ka = cat8(ld8(kp + off), ld8(kp + off + 16));
        c = wmma_bf16(ka, qf[s], c);
      }
      st[t] = c;
    }

    // Online softmax along keys (= rows of S^T = regs + xor-16 partner lane).
    float bm = -3.0e38f;
#pragma unroll
    for (int r = 0; r < 8; ++r) {
      bm = fmaxf(bm, st[0][r]);
      bm = fmaxf(bm, st[1][r]);
    }
    bm = fmaxf(bm, __shfl_xor(bm, 16, 32));
    float mnew = fmaxf(mrun, bm);
    float alpha = __expf(mrun - mnew);
    float p0[8], p1[8], psum = 0.0f;
#pragma unroll
    for (int r = 0; r < 8; ++r) {
      p0[r] = __expf(st[0][r] - mnew);
      p1[r] = __expf(st[1][r] - mnew);
      psum += p0[r] + p1[r];
    }
    psum += __shfl_xor(psum, 16, 32);
    lrun = lrun * alpha + psum;
    mrun = mnew;

    // Rescale O: O-tile reg r holds query row 8hg+r -> fetch that query's
    // alpha from lane (8hg+r) (state replicated in lanes q and q+16).
#pragma unroll
    for (int r = 0; r < 8; ++r) {
      float ar = __shfl(alpha, hg * 8 + r, 32);
#pragma unroll
      for (int dt = 0; dt < 4; ++dt) o[dt][r] *= ar;
    }

    // P fragment (A-operand, 16q x 32k): lane=query(lr), elems 0..7 = keys
    // 8hg+0..7 (tile0), elems 8..15 = keys 16+8hg+0..7 (tile1). Exactly st[].
    v16bf pf;
#pragma unroll
    for (int r = 0; r < 8; ++r) {
      pf[r] = (bf16)p0[r];
      pf[r + 8] = (bf16)p1[r];
    }

    // O += P @ V ; vT rows are contiguous in key -> direct B-operand loads.
#pragma unroll
    for (int dt = 0; dt < 4; ++dt) {
      int off = (dt * 16 + lr) * Nc + kb0 + hg * 8;
      v16bf vb = cat8(ld8(vp + off), ld8(vp + off + 16));
      o[dt] = wmma_bf16(pf, vb, o[dt]);
    }
  }

  // Normalize: softmax row / sqrt(E) = exp-sum / (32 * l).
  float f = 1.0f / (32.0f * lrun);
  const int b = bh >> 4, h = bh & 15;
#pragma unroll
  for (int r = 0; r < 8; ++r) {
    float fr = __shfl(f, hg * 8 + r, 32);
#pragma unroll
    for (int dt = 0; dt < 4; ++dt) {
      int qrow = qt + hg * 8 + r;
      int d = dt * 16 + lr;
      attn[((size_t)(b * Nc + qrow)) * Ec + h * Dc + d] = (bf16)(o[dt][r] * fr);
    }
  }
}

// ---------------------------------------------------------------------------
// Host-side orchestration
// ---------------------------------------------------------------------------
extern "C" void kernel_launch(void* const* d_in, const int* in_sizes, int n_in,
                              void* d_out, int out_size, void* d_ws,
                              size_t ws_size, hipStream_t stream) {
  (void)in_sizes; (void)n_in; (void)out_size; (void)ws_size;
  const float* x  = (const float*)d_in[0];
  const float* Wq = (const float*)d_in[1];
  const float* bq = (const float*)d_in[2];
  const float* Wk = (const float*)d_in[3];
  const float* bk = (const float*)d_in[4];
  const float* Wv = (const float*)d_in[5];
  const float* bv = (const float*)d_in[6];
  const float* Wo = (const float*)d_in[7];
  const float* bo = (const float*)d_in[8];

  char* ws = (char*)d_ws;
  const size_t xElems = (size_t)Mrows * Ec;        // 4,194,304
  const size_t wElems = (size_t)Ec * Ec;           // 1,048,576
  const size_t hElems = (size_t)Bc * Hc * Nc * Dc; // 4,194,304

  bf16* xb  = (bf16*)(ws);
  bf16* wqb = (bf16*)(ws + 2 * xElems);
  bf16* wkb = (bf16*)(ws + 2 * xElems + 2 * wElems);
  bf16* wvb = (bf16*)(ws + 2 * xElems + 4 * wElems);
  bf16* wob = (bf16*)(ws + 2 * xElems + 6 * wElems);
  bf16* qh  = (bf16*)(ws + 2 * xElems + 8 * wElems);
  bf16* kh  = (bf16*)(ws + 2 * xElems + 8 * wElems + 2 * hElems);
  bf16* vth = (bf16*)(ws + 2 * xElems + 8 * wElems + 4 * hElems);
  bf16* atb = (bf16*)(ws + 2 * xElems + 8 * wElems + 6 * hElems);

  // 1) f32 -> bf16
  cvt_bf16_kernel<<<(int)(xElems / 256), 256, 0, stream>>>(x, xb, (int)xElems);
  cvt_bf16_kernel<<<(int)(wElems / 256), 256, 0, stream>>>(Wq, wqb, (int)wElems);
  cvt_bf16_kernel<<<(int)(wElems / 256), 256, 0, stream>>>(Wk, wkb, (int)wElems);
  cvt_bf16_kernel<<<(int)(wElems / 256), 256, 0, stream>>>(Wv, wvb, (int)wElems);
  cvt_bf16_kernel<<<(int)(wElems / 256), 256, 0, stream>>>(Wo, wob, (int)wElems);

  // 2) QKV projections (x @ W^T + b) -> head layouts
  dim3 gg(Ec / 128, Mrows / 128);
  wmma_gemm_bias<0><<<gg, 256, 0, stream>>>(xb, wqb, bq, qh, Mrows, Ec, Ec);
  wmma_gemm_bias<0><<<gg, 256, 0, stream>>>(xb, wkb, bk, kh, Mrows, Ec, Ec);
  wmma_gemm_bias<1><<<gg, 256, 0, stream>>>(xb, wvb, bv, vth, Mrows, Ec, Ec);

  // 3) attention (softmax then 1/sqrt(E)), merged-head bf16 result
  flash_attn_kernel<<<Bc * Hc * (Nc / 64), 128, 0, stream>>>(qh, kh, vth, atb);

  // 4) output projection -> f32 d_out
  wmma_gemm_bias<2><<<gg, 256, 0, stream>>>(atb, wob, bo, (float*)d_out,
                                            Mrows, Ec, Ec);
}
